// SimpleMambaClassifier_18451179504229
// MI455X (gfx1250) — compile-verified, matrix operand order checked by reading, not run on hardware
//
#include <hip/hip_runtime.h>

#define D_MODEL   64
#define NUM_CLASSES 2
#define D_STATE   16
#define D_CONV    4
#define D_INNER   128
#define DT_RANK   4
#define B_SZ      16
#define SEQ       4096
#define NTOK      (B_SZ * SEQ)   // 65536 tokens
#define XD        36             // dt_rank + 2*d_state
#define CHUNK     64             // scan time-chunk staged in LDS
#define NSEG      16             // time segments for parallel scan
#define SEGLEN    (SEQ / NSEG)   // 256
#define NCH       (SEGLEN / CHUNK) // 4 chunks per segment

typedef __attribute__((ext_vector_type(16))) _Float16 v16h;
typedef __attribute__((ext_vector_type(8)))  float    v8f;

#if defined(__HIP_DEVICE_COMPILE__) && __has_builtin(__builtin_amdgcn_tensor_load_to_lds) && __has_builtin(__builtin_amdgcn_s_wait_tensorcnt)
#define HAVE_TDM 1
#else
#define HAVE_TDM 0
#endif

#if HAVE_TDM
typedef unsigned int v4u __attribute__((ext_vector_type(4)));
typedef int          v8i __attribute__((ext_vector_type(8)));
typedef int          v4i __attribute__((ext_vector_type(4)));

// Issue a TDM load of `nelem` contiguous f32 (1-D tile) from global into LDS.
// D# layout per CDNA5 ISA: group0 = {flags, lds_addr, global_addr, type},
// group1 = {data_size=4B, tensor_dim0 = tile_dim0 = nelem, dims1/2 unused}.
// This toolchain exposes the 6-arg builtin (clang-23 / therock-10.0 flavor):
//   (uint32x4, int32x8, int32x4, int32x4, int32x8, i32 cpol) — trailing groups zero-filled.
__device__ __forceinline__ void tdm_load_1d(unsigned lds_addr, const float* gsrc, unsigned nelem) {
  const unsigned long long ga = (unsigned long long)(size_t)gsrc;
  v4u g0;
  g0[0] = 1u;                                   // count=1, user descriptor
  g0[1] = lds_addr;                             // LDS byte address (bits 63:32)
  g0[2] = (unsigned)ga;                         // global_addr[31:0]
  g0[3] = (unsigned)(ga >> 32) | (2u << 30);    // global_addr[56:32] | type=2 ("image")
  v8i g1;
  g1[0] = (int)(2u << 16);                      // workgroup_mask=0, data_size=2 (4B)
  g1[1] = (int)((nelem & 0xffffu) << 16);       // tensor_dim0[15:0] @ bits 63:48
  g1[2] = (int)(nelem >> 16);                   // tensor_dim0[31:16]; tensor_dim1[15:0]=0
  g1[3] = (int)((nelem & 0xffffu) << 16);       // tile_dim0 = nelem @ bits 127:112
  g1[4] = 0;                                    // tile_dim1=0 (unused), tile_dim2=0
  g1[5] = (int)nelem;                           // tensor_dim0_stride[31:0] (benign for 1-D)
  g1[6] = 0;
  g1[7] = 0;
  const v4i zz4 = {0, 0, 0, 0};
  const v8i zz8 = {0, 0, 0, 0, 0, 0, 0, 0};
  __builtin_amdgcn_tensor_load_to_lds(g0, g1, zz4, zz4, zz8, 0);
}
#endif

__device__ __forceinline__ v16h frag_zero() {
  v16h v;
#pragma unroll
  for (int i = 0; i < 16; ++i) v[i] = (_Float16)0.f;
  return v;
}

// 16x32 A/B fragment (WMMA f16 layout) from f16 row-major storage.
// Lanes 0-15: row=lane, K = {0..7} U {16..23}; lanes 16-31: row=lane-16, K = {8..15} U {24..31}.
__device__ __forceinline__ v16h frag_ld_f16(const _Float16* tile, int ld, int lane) {
  const int h = (lane >> 4) & 1;
  const int r = lane & 15;
  const _Float16* q = tile + r * ld + h * 8;
  v16h v;
#pragma unroll
  for (int i = 0; i < 8; ++i) { v[i] = q[i]; v[8 + i] = q[16 + i]; }
  return v;
}

// Same layout from f32 storage with on-the-fly f16 conversion; rows >= rows_valid give 0.
__device__ __forceinline__ v16h frag_ld_f32(const float* tile, int ld, int lane, int rows_valid) {
  const int h = (lane >> 4) & 1;
  const int r = lane & 15;
  v16h v = frag_zero();
  if (r < rows_valid) {
    const float* q = tile + r * ld + h * 8;
#pragma unroll
    for (int i = 0; i < 8; ++i) { v[i] = (_Float16)q[i]; v[8 + i] = (_Float16)q[16 + i]; }
  }
  return v;
}

// ---------------- Kernel 1: in_proj  xz = x @ Wᵀ  (65536x64 @ 64x256) -> u|z (f16) ----
__global__ void k_in_proj(const float* __restrict__ x, const float* __restrict__ w,
                          _Float16* __restrict__ u, _Float16* __restrict__ z) {
  const int lane = threadIdx.x & 31;
  const int wave = threadIdx.x >> 5;
  const int tile = blockIdx.x * 8 + wave;   // 65536 tiles total
  const int mt = tile >> 4;                 // 4096 M-tiles
  const int nt = tile & 15;                 // 16  N-tiles
  const int m0 = mt * 16, n0 = nt * 16;
  v8f acc = {};
#pragma unroll
  for (int k0 = 0; k0 < D_MODEL; k0 += 32) {
    v16h a = frag_ld_f32(x + (size_t)m0 * D_MODEL + k0, D_MODEL, lane, 16);
    v16h b = frag_ld_f32(w + (size_t)n0 * D_MODEL + k0, D_MODEL, lane, 16);
    acc = __builtin_amdgcn_wmma_f32_16x16x32_f16(false, a, false, b, (short)0, acc, false, false);
  }
  const int h  = (lane >> 4) & 1;
  const int nl = lane & 15;
  // n0 is tile-uniform and the u/z boundary (128) is a multiple of 16, so the
  // destination select is wave-uniform -> no per-element EXEC juggling.
  _Float16* dst = (n0 < D_INNER) ? (u + n0 + nl) : (z + (n0 - D_INNER) + nl);
#pragma unroll
  for (int r = 0; r < 8; ++r) {
    const int m = m0 + r + h * 8;
    dst[(size_t)m * D_INNER] = (_Float16)acc[r];
  }
}

// ---------------- Kernel 2: causal depthwise conv1d (K=4) + bias + SiLU ---------------
__global__ void k_conv_silu(const _Float16* __restrict__ u, const float* __restrict__ cw,
                            const float* __restrict__ cb, _Float16* __restrict__ uc) {
  const int idx = blockIdx.x * blockDim.x + threadIdx.x;  // NTOK * D_INNER
  const int d   = idx & (D_INNER - 1);
  const int tok = idx >> 7;
  const int t   = tok & (SEQ - 1);
  float acc = cb[d];
#pragma unroll
  for (int j = 0; j < D_CONV; ++j) {
    const int tt = t - (D_CONV - 1) + j;
    if (tt >= 0)
      acc += (float)u[(size_t)(tok - (D_CONV - 1) + j) * D_INNER + d] * cw[d * D_CONV + j];
  }
  const float s = acc / (1.f + __expf(-acc));  // SiLU
  uc[idx] = (_Float16)s;
}

// ---------------- Kernel 3: x_proj  x_dbl = uc @ Wᵀ  (65536x128 @ 128x36, N pad 48) ---
__global__ void k_x_proj(const _Float16* __restrict__ uc, const float* __restrict__ w,
                         float* __restrict__ xdbl) {
  const int lane = threadIdx.x & 31;
  const int wave = threadIdx.x >> 5;
  const int tile = blockIdx.x * 8 + wave;   // 12288 tiles
  const int mt = tile / 3;
  const int nt = tile % 3;
  const int m0 = mt * 16, n0 = nt * 16;
  const int nvalid = (nt == 2) ? (XD - 32) : 16;
  v8f acc = {};
#pragma unroll
  for (int k0 = 0; k0 < D_INNER; k0 += 32) {
    v16h a = frag_ld_f16(uc + (size_t)m0 * D_INNER + k0, D_INNER, lane);
    v16h b = frag_ld_f32(w + (size_t)n0 * D_INNER + k0, D_INNER, lane, nvalid);
    acc = __builtin_amdgcn_wmma_f32_16x16x32_f16(false, a, false, b, (short)0, acc, false, false);
  }
  const int h = (lane >> 4) & 1;
  const int n = n0 + (lane & 15);
  if (n < XD) {
#pragma unroll
    for (int r = 0; r < 8; ++r) {
      const int m = m0 + r + h * 8;
      xdbl[(size_t)m * XD + n] = acc[r];
    }
  }
}

// ---------------- Kernel 4a: segmented selective scan (pass 1) ------------------------
// One block per (batch, segment); lane = d_inner channel.  For segment with unknown
// initial state h0 (linear recurrence):
//   h_t = P_t ⊙ h0 + part_t,   P_t = Π dA,  part_t = dA·part + dt·u·B_t
//   Σ_t silu(z)·y_t = aA + aB·h0  with aA = Σ silu(z)(C·part_t + u·D),
//                                   aB[s] = Σ silu(z)·C_t[s]·P_t[s]
// dt/B/C rows stream through double-buffered LDS via the Tensor Data Mover.
__global__ void __launch_bounds__(D_INNER)
k_scan_seg(const float* __restrict__ xdbl, const _Float16* __restrict__ uc,
           const _Float16* __restrict__ z,
           const float* __restrict__ dtw, const float* __restrict__ dtb,
           const float* __restrict__ alog, const float* __restrict__ Dp,
           float* __restrict__ PendO, float* __restrict__ partO,
           float* __restrict__ aBO, float* __restrict__ aAO) {
  const int b   = blockIdx.x / NSEG;
  const int seg = blockIdx.x % NSEG;
  const int d   = threadIdx.x;
  __shared__ float sh[2][CHUNK * XD];

  const float w0 = dtw[d * DT_RANK + 0], w1 = dtw[d * DT_RANK + 1];
  const float w2 = dtw[d * DT_RANK + 2], w3 = dtw[d * DT_RANK + 3];
  const float bias = dtb[d], Dd = Dp[d];
  float As[D_STATE], P[D_STATE], part[D_STATE], aB[D_STATE];
#pragma unroll
  for (int s = 0; s < D_STATE; ++s) {
    As[s] = -__expf(alog[d * D_STATE + s]);
    P[s] = 1.f; part[s] = 0.f; aB[s] = 0.f;
  }
  float aA = 0.f;

  const size_t tok0 = (size_t)b * SEQ + (size_t)seg * SEGLEN;
  const float*    xseg = xdbl + tok0 * XD;
  const _Float16* ucb  = uc + tok0 * D_INNER + d;
  const _Float16* zb   = z  + tok0 * D_INNER + d;

#if HAVE_TDM
  if (threadIdx.x < 32)
    tdm_load_1d((unsigned)(size_t)&sh[0][0], xseg, CHUNK * XD);
#endif
  int buf = 0;
  for (int c = 0; c < NCH; ++c) {
#if HAVE_TDM
    if (threadIdx.x < 32) __builtin_amdgcn_s_wait_tensorcnt(0);
    __syncthreads();                       // buf ready; previous compute on buf^1 done
    if (c + 1 < NCH && threadIdx.x < 32)
      tdm_load_1d((unsigned)(size_t)&sh[buf ^ 1][0],
                  xseg + (size_t)(c + 1) * CHUNK * XD, CHUNK * XD);
#else
    for (int i = threadIdx.x; i < CHUNK * XD; i += D_INNER)
      sh[buf][i] = xseg[(size_t)c * CHUNK * XD + i];
    __syncthreads();
#endif
    const float* shc = sh[buf];
    for (int ct = 0; ct < CHUNK; ++ct) {
      const int t = c * CHUNK + ct;
      const float* sr = shc + ct * XD;     // [dt0..3 | B0..15 | C0..15]
      const float dtv = w0 * sr[0] + w1 * sr[1] + w2 * sr[2] + w3 * sr[3] + bias;
      const float dt  = (dtv > 20.f) ? dtv : log1pf(__expf(dtv));   // softplus
      const float uv  = (float)ucb[(size_t)t * D_INNER];
      const float zv  = (float)zb[(size_t)t * D_INNER];
      const float sv  = zv / (1.f + __expf(-zv));   // silu(z)
      const float dtu = dt * uv;
      float ya = uv * Dd;
#pragma unroll
      for (int s = 0; s < D_STATE; ++s) {
        const float dA = __expf(dt * As[s]);
        P[s]   *= dA;
        part[s] = dA * part[s] + dtu * sr[DT_RANK + s];
        ya     += part[s] * sr[DT_RANK + D_STATE + s];
        aB[s]  += sv * sr[DT_RANK + D_STATE + s] * P[s];
      }
      aA += sv * ya;
    }
#if !HAVE_TDM
    __syncthreads();
#endif
    buf ^= 1;
  }

  const size_t base = ((size_t)(b * NSEG + seg) * D_STATE) * D_INNER + d;
#pragma unroll
  for (int s = 0; s < D_STATE; ++s) {
    PendO[base + (size_t)s * D_INNER] = P[s];
    partO[base + (size_t)s * D_INNER] = part[s];
    aBO  [base + (size_t)s * D_INNER] = aB[s];
  }
  aAO[(size_t)(b * NSEG + seg) * D_INNER + d] = aA;
}

// ---------------- Kernel 4b: chain segments (pass 2) ----------------------------------
__global__ void __launch_bounds__(D_INNER)
k_scan_combine(const float* __restrict__ Pend, const float* __restrict__ partE,
               const float* __restrict__ aBv, const float* __restrict__ aAv,
               float* __restrict__ ysum) {
  const int b = blockIdx.x;
  const int d = threadIdx.x;
  float h[D_STATE];
#pragma unroll
  for (int s = 0; s < D_STATE; ++s) h[s] = 0.f;
  float tot = 0.f;
  for (int seg = 0; seg < NSEG; ++seg) {
    const size_t base = ((size_t)(b * NSEG + seg) * D_STATE) * D_INNER + d;
    float t = aAv[(size_t)(b * NSEG + seg) * D_INNER + d];
#pragma unroll
    for (int s = 0; s < D_STATE; ++s) t += aBv[base + (size_t)s * D_INNER] * h[s];
    tot += t;
#pragma unroll
    for (int s = 0; s < D_STATE; ++s)
      h[s] = Pend[base + (size_t)s * D_INNER] * h[s] + partE[base + (size_t)s * D_INNER];
  }
  ysum[b * D_INNER + d] = tot * (1.f / SEQ);   // time-mean (pool before out_proj)
}

// ---------------- Kernel 5: head  pooled = ysum @ Woᵀ ; logits = pooled @ Wcᵀ + bc ----
__global__ void k_head(const float* __restrict__ ysum, const float* __restrict__ wo,
                       const float* __restrict__ wc, const float* __restrict__ bc,
                       float* __restrict__ out) {
  const int lane = threadIdx.x & 31;
  __shared__ _Float16 pooled[16 * D_MODEL];
  const int h  = (lane >> 4) & 1;
  const int nl = lane & 15;

#pragma unroll
  for (int nt = 0; nt < D_MODEL / 16; ++nt) {
    v8f acc = {};
#pragma unroll
    for (int k0 = 0; k0 < D_INNER; k0 += 32) {
      v16h a = frag_ld_f32(ysum + k0, D_INNER, lane, 16);
      v16h b = frag_ld_f32(wo + (size_t)(nt * 16) * D_INNER + k0, D_INNER, lane, 16);
      acc = __builtin_amdgcn_wmma_f32_16x16x32_f16(false, a, false, b, (short)0, acc, false, false);
    }
#pragma unroll
    for (int r = 0; r < 8; ++r)
      pooled[(r + h * 8) * D_MODEL + nt * 16 + nl] = (_Float16)acc[r];
  }
  __syncthreads();

  v8f acc = {};
#pragma unroll
  for (int k0 = 0; k0 < D_MODEL; k0 += 32) {
    v16h a = frag_ld_f16(pooled + k0, D_MODEL, lane);
    v16h b = frag_ld_f32(wc + k0, D_MODEL, lane, NUM_CLASSES);
    acc = __builtin_amdgcn_wmma_f32_16x16x32_f16(false, a, false, b, (short)0, acc, false, false);
  }
  if (nl < NUM_CLASSES) {
    const float bb = bc[nl];
#pragma unroll
    for (int r = 0; r < 8; ++r) {
      const int m = r + h * 8;                 // batch index 0..15
      out[m * NUM_CLASSES + nl] = acc[r] + bb;
    }
  }
}

extern "C" void kernel_launch(void* const* d_in, const int* in_sizes, int n_in,
                              void* d_out, int out_size, void* d_ws, size_t ws_size,
                              hipStream_t stream) {
  const float* x    = (const float*)d_in[0];
  const float* wi   = (const float*)d_in[1];
  const float* cw   = (const float*)d_in[2];
  const float* cb   = (const float*)d_in[3];
  const float* wx   = (const float*)d_in[4];
  const float* dtw  = (const float*)d_in[5];
  const float* dtb  = (const float*)d_in[6];
  const float* alog = (const float*)d_in[7];
  const float* Dp   = (const float*)d_in[8];
  const float* wo   = (const float*)d_in[9];
  const float* wc   = (const float*)d_in[10];
  const float* bc   = (const float*)d_in[11];
  float* out = (float*)d_out;

  // Workspace layout (~66 MB total, fits in L2):
  const size_t segN = (size_t)B_SZ * NSEG * D_STATE * D_INNER;   // 524288
  _Float16* u  = (_Float16*)d_ws;                       // 65536*128 f16
  _Float16* z  = u  + (size_t)NTOK * D_INNER;           // 65536*128 f16
  _Float16* uc = z  + (size_t)NTOK * D_INNER;           // 65536*128 f16
  float* xdbl  = (float*)(uc + (size_t)NTOK * D_INNER); // 65536*36  f32
  float* ysum  = xdbl + (size_t)NTOK * XD;              // 16*128    f32
  float* Pend  = ysum + B_SZ * D_INNER;                 // segment cumprod endpoints
  float* partE = Pend + segN;                           // segment particular solutions
  float* aBv   = partE + segN;                          // state-coupling reductions
  float* aAv   = aBv + segN;                            // scalar reductions

  k_in_proj <<<dim3((NTOK / 16) * (2 * D_INNER / 16) / 8), dim3(256), 0, stream>>>(x, wi, u, z);
  k_conv_silu<<<dim3(NTOK * D_INNER / 256), dim3(256), 0, stream>>>(u, cw, cb, uc);
  k_x_proj  <<<dim3((NTOK / 16) * 3 / 8), dim3(256), 0, stream>>>(uc, wx, xdbl);
  k_scan_seg<<<dim3(B_SZ * NSEG), dim3(D_INNER), 0, stream>>>(xdbl, uc, z, dtw, dtb, alog, Dp,
                                                              Pend, partE, aBv, aAv);
  k_scan_combine<<<dim3(B_SZ), dim3(D_INNER), 0, stream>>>(Pend, partE, aBv, aAv, ysum);
  k_head    <<<dim3(1), dim3(32), 0, stream>>>(ysum, wo, wc, bc, out);
}
